// DDPM_326417514870
// MI455X (gfx1250) — matrix operand order repaired and checked by reading, not compiled, not run on hardware
//
#include <hip/hip_runtime.h>
#include <math.h>

typedef unsigned short u16t;
typedef unsigned int   u32t;
typedef __bf16 bf16t;
typedef bf16t v16bf __attribute__((ext_vector_type(16)));
typedef float v8f   __attribute__((ext_vector_type(8)));
typedef int   v4i_t __attribute__((ext_vector_type(4)));

struct bfrag { uint4 a, b; };   // 32 bytes = 16 bf16 per lane

#if defined(__has_builtin)
#if __has_builtin(__builtin_amdgcn_global_load_async_to_lds_b128)
#define HAS_ASYNC_LDS 1
#else
#define HAS_ASYNC_LDS 0
#endif
#if __has_builtin(__builtin_amdgcn_s_wait_asynccnt)
#define HAS_WAIT_ASYNC_BUILTIN 1
#else
#define HAS_WAIT_ASYNC_BUILTIN 0
#endif
#else
#define HAS_ASYNC_LDS 0
#define HAS_WAIT_ASYNC_BUILTIN 0
#endif

// 16B memory -> LDS copy; async (ASYNCcnt-tracked, no VGPR round-trip) when
// the toolchain exposes the gfx1250 builtin, else plain load+store.
// Builtin signature (from hipcc diagnostic): (v4i addrspace(1)*, v4i
// addrspace(3)*, imm offset, imm cpol).
__device__ __forceinline__ void async_copy16(const u16t* g, u16t* l) {
#if HAS_ASYNC_LDS
  __builtin_amdgcn_global_load_async_to_lds_b128(
      (__attribute__((address_space(1))) v4i_t*)(g),
      (__attribute__((address_space(3))) v4i_t*)(l), 0, 0);
#else
  *(uint4*)l = *(const uint4*)g;
#endif
}
__device__ __forceinline__ void wait_async() {
#if HAS_ASYNC_LDS
#if HAS_WAIT_ASYNC_BUILTIN
  __builtin_amdgcn_s_wait_asynccnt(0);
#else
  asm volatile("s_wait_asynccnt 0" ::: "memory");
#endif
#endif
}

__device__ __forceinline__ v16bf frag_cast(bfrag f) {
  return __builtin_bit_cast(v16bf, f);
}
__device__ __forceinline__ u16t f2bf(float f) {
  u32t u = __builtin_bit_cast(u32t, f);
  u32t r = u + 0x7FFFu + ((u >> 16) & 1u);   // round-to-nearest-even
  return (u16t)(r >> 16);
}
__device__ __forceinline__ float bf2f(u16t h) {
  u32t u = ((u32t)h) << 16;
  return __builtin_bit_cast(float, u);
}
__device__ __forceinline__ float mishf(float x) {
  float sp = log1pf(__expf(x));
  return x * tanhf(sp);
}
__device__ __forceinline__ v8f wmma_bf16(v16bf a, v16bf b, v8f c) {
  return __builtin_amdgcn_wmma_f32_16x16x32_bf16(false, a, false, b, (short)0, c,
                                                 false, false);
}

// ---------------------------------------------------------------------------
// Repack f32 weight [K,N] row-major into bf16 WMMA B-fragment order.
// Fragment block (nt,kk) covers cols nt*16..+15, k kk*32..+31.
// Within a block: lane l (0..31): n = nt*16 + (l&15), k-half = l>>4;
// element e (0..15): k = kk*32 + (l>>4)*16 + e  (16 contiguous k per lane).
// Flat offset = ((nt*(K/32)+kk)*32 + lane)*16 + e  -> each lane reads its
// fragment as two contiguous b128 loads; a 32x16 k-tile is 1KB contiguous.
// ---------------------------------------------------------------------------
__global__ void pack_w_kernel(const float* __restrict__ W, u16t* __restrict__ Wp,
                              int K, int N) {
  long long flat = (long long)blockIdx.x * 256 + threadIdx.x;
  long long total = (long long)K * N;
  if (flat >= total) return;
  int e    = (int)(flat & 15);
  int lane = (int)((flat >> 4) & 31);
  long long blk = flat >> 9;
  int kblocks = K >> 5;
  int kk = (int)(blk % kblocks);
  long long nt = blk / kblocks;
  int n = (int)(nt * 16 + (lane & 15));
  int k = kk * 32 + (lane >> 4) * 16 + e;
  Wp[flat] = f2bf(W[(long long)k * N + n]);
}

__global__ void cvt_bf16_kernel(const float* __restrict__ x, u16t* __restrict__ y, int n) {
  int i = blockIdx.x * 256 + threadIdx.x;
  if (i < n) y[i] = f2bf(x[i]);
}

// Schedule table: sched[t] = {beta/sqrt(1-ab), 1/sqrt(alpha), sigma, 0}
__global__ void sched_kernel(float* __restrict__ sched, float* __restrict__ zeros) {
  int tid = threadIdx.x;
  if (tid < 128) zeros[tid] = 0.0f;
  if (tid == 0) {
    float ab = 1.0f;
    for (int t = 0; t < 50; ++t) {
      float beta = 1e-4f + (0.02f - 1e-4f) * ((float)t / 49.0f);
      float alpha = 1.0f - beta;
      float ab_prev = ab;
      ab *= alpha;
      float c1 = beta / sqrtf(1.0f - ab);
      float c2 = 1.0f / sqrtf(alpha);
      float s2 = fmaxf(beta * (1.0f - ab_prev) / (1.0f - ab), 1e-20f);
      float sg = (t > 0) ? sqrtf(s2) : 0.0f;
      sched[t * 4 + 0] = c1; sched[t * 4 + 1] = c2;
      sched[t * 4 + 2] = sg; sched[t * 4 + 3] = 0.0f;
    }
  }
}

// t-embedding MLP once for all 50 steps (t is batch-invariant).
// block = one timestep, 256 threads.
__global__ void tff_kernel(const float* __restrict__ t_w1, const float* __restrict__ t_b1,
                           const float* __restrict__ t_w2, const float* __restrict__ t_b2,
                           u16t* __restrict__ tff) {
  __shared__ float pf[128];
  __shared__ float h[256];
  const int t = blockIdx.x, tid = threadIdx.x;
  if (tid < 64) {
    float fr = __expf((float)tid * (-9.210340371976184f / 64.0f)); // (1e-4)^(j/64)
    float x = (float)t * fr;
    pf[tid] = __cosf(x);
    pf[tid + 64] = __sinf(x);
  }
  __syncthreads();
  {
    float s = t_b1[tid];
    for (int i = 0; i < 128; ++i) s += pf[i] * t_w1[i * 256 + tid];
    h[tid] = mishf(s);
  }
  __syncthreads();
  if (tid < 128) {
    float s = t_b2[tid];
    for (int j = 0; j < 256; ++j) s += h[j] * t_w2[j * 128 + tid];
    tff[t * 128 + tid] = f2bf(s);
  }
}

// xin[b, 0:128] = bf16(xt[b]); xin[b, 128:256] = t_ff[t]
__global__ void build_xin_kernel(const float* __restrict__ xt, const u16t* __restrict__ tff,
                                 int t, u16t* __restrict__ xin) {
  int i = blockIdx.x * 256 + threadIdx.x;
  int row = i >> 8, c = i & 255;
  xin[i] = (c < 128) ? f2bf(xt[row * 128 + c]) : tff[t * 128 + (c - 128)];
}

// ---------------------------------------------------------------------------
// Generic GEMM:  out[M=1024, N] = act(A[M,K](bf16) @ Wp(packed bf16) + bias)
// Block: 128 thr = 4 waves, tile 64(M) x 64(N); each wave: 16 rows x 4
// independent WMMA chains. 64-row M-tiles give 256 blocks for the 1024x1024
// layers (fills a ~256-WGP part with 2 waves/SIMD). A is double-buffered in
// LDS and staged with async-to-LDS copies; B streams from packed-fragment
// global memory, fully coalesced.
// ---------------------------------------------------------------------------
template <bool MISH, bool OUTF32>
__global__ __launch_bounds__(128) void gemm_kernel(
    const u16t* __restrict__ A, const u16t* __restrict__ Wp,
    const float* __restrict__ bias, void* __restrict__ out,
    int K, int N, int ldo, int col_off) {
  __shared__ u16t As[2][64 * 40];
  const int tid = threadIdx.x;
  const int wave = tid >> 5, lane = tid & 31;
  const int row = lane & 15, half = lane >> 4;
  const int m0 = blockIdx.x * 64;
  const int ntile0 = blockIdx.y * 4;
  const int kblocks = K >> 5;
  const size_t nstride = (size_t)kblocks * 512;

  v8f acc[4];
  for (int n = 0; n < 4; ++n)
    for (int q = 0; q < 8; ++q) acc[n][q] = 0.0f;

  // stage k-tile kk into buffer buf: 64 rows x 32 k (bf16), 256 x 16B
  auto stage = [&](int kk, int buf) {
    for (int idx = tid; idx < 256; idx += 128) {
      int r = idx >> 2, c = idx & 3;
      async_copy16(A + (size_t)(m0 + r) * K + kk * 32 + c * 8,
                   &As[buf][r * 40 + c * 8]);
    }
  };

  stage(0, 0);
  wait_async();
  __syncthreads();

  int buf = 0;
  for (int kk = 0; kk < kblocks; ++kk) {
    if (kk + 1 < kblocks) stage(kk + 1, buf ^ 1);   // overlap with compute
    // A fragment: lane row=lane&15, contiguous k-run
    // {half*16 .. +15} (matches packed B)
    bfrag af;
    const u16t* ap = &As[buf][(wave * 16 + row) * 40 + half * 16];
    af.a = *(const uint4*)(ap);
    af.b = *(const uint4*)(ap + 8);
    v16bf av = frag_cast(af);
    const u16t* bp = Wp + ((size_t)ntile0 * kblocks + kk) * 512 + (size_t)lane * 16;
    for (int nsub = 0; nsub < 4; ++nsub) {
      bfrag bf_;
      bf_.a = *(const uint4*)(bp);
      bf_.b = *(const uint4*)(bp + 8);
      acc[nsub] = wmma_bf16(av, frag_cast(bf_), acc[nsub]);
      bp += nstride;
    }
    wait_async();
    __syncthreads();
    buf ^= 1;
  }
  for (int nsub = 0; nsub < 4; ++nsub) {
    int col = (ntile0 + nsub) * 16 + row;
    float bv = bias[col];
    for (int q = 0; q < 8; ++q) {
      int orow = m0 + wave * 16 + half * 8 + q;   // D: M = q + 8*half
      float v = acc[nsub][q] + bv;
      if (MISH) v = mishf(v);
      if (OUTF32)
        ((float*)out)[(size_t)orow * ldo + col_off + col] = v;
      else
        ((u16t*)out)[(size_t)orow * ldo + col_off + col] = f2bf(v);
    }
  }
}

// ---------------------------------------------------------------------------
// Fused ζ-output layer + einsum, f-split for occupancy:
// spart[fc][b,s] = Σ_{f in chunk fc} psi[b,f] * (h1[b,:]·W2[:, f*128+s])
// Grid (32, 2, 4) = 256 blocks. Block: 32 b-rows x 64 s-cols x 64 f-values,
// 8 waves (2 b-subs x 4 s-subs). h1 tile (32x1032 bf16, padded) is
// async-staged to LDS and reused across the 64 f iterations; psiT chunk also
// in LDS. W2 streams from L2 in packed fragment order. The k-reduction runs
// as TWO independent WMMA chains (even/odd k-tiles); the z tile never touches
// memory: scaled by psi and folded into the score accumulator immediately.
// Partials land in disjoint slices (bitwise-deterministic, no atomics).
// ---------------------------------------------------------------------------
__global__ __launch_bounds__(256) void fused_score_kernel(
    const u16t* __restrict__ h1, const u16t* __restrict__ w2p,
    const u16t* __restrict__ psib, float* __restrict__ spart) {
  extern __shared__ u16t sm[];
  u16t* Ah = sm;               // [32][1032] padded h1 tile
  u16t* Ps = sm + 32 * 1032;   // psiT chunk [64][32]
  const int tid = threadIdx.x;
  const int wave = tid >> 5, lane = tid & 31;
  const int row = lane & 15, half = lane >> 4;
  const int bsub = wave & 1, ssub = wave >> 1;
  const int b0 = blockIdx.x * 32;
  const int stile = blockIdx.y * 4 + ssub;   // 0..7
  const int f0 = blockIdx.z * 64;            // f chunk

  for (int idx = tid; idx < 4096; idx += 256) {   // h1 tile: 32 rows x 128 x 16B
    int r = idx >> 7, c = idx & 127;
    async_copy16(h1 + (size_t)(b0 + r) * 1024 + c * 8, Ah + r * 1032 + c * 8);
  }
  for (int idx = tid; idx < 2048; idx += 256) {   // psi chunk transpose into LDS
    int fo = idx >> 5, r = idx & 31;
    Ps[fo * 32 + r] = psib[(size_t)(b0 + r) * 256 + f0 + fo];
  }
  wait_async();
  __syncthreads();

  v8f score;
  for (int q = 0; q < 8; ++q) score[q] = 0.0f;
  const int arow = bsub * 16 + row;
  const u16t* apbase = &Ah[arow * 1032 + half * 16];

  for (int fo = 0; fo < 64; ++fo) {
    v8f z0, z1;
    for (int q = 0; q < 8; ++q) { z0[q] = 0.0f; z1[q] = 0.0f; }
    const int nt = (f0 + fo) * 8 + stile;                 // flat col tile of W2
    const u16t* bp = w2p + (size_t)nt * 32 * 512 + (size_t)lane * 16;
    const u16t* ap = apbase;
    for (int kk = 0; kk < 32; kk += 2) {                  // two parallel chains
      bfrag af0, bf0, af1, bf1;
      af0.a = *(const uint4*)(ap);
      af0.b = *(const uint4*)(ap + 8);
      bf0.a = *(const uint4*)(bp);
      bf0.b = *(const uint4*)(bp + 8);
      af1.a = *(const uint4*)(ap + 32);
      af1.b = *(const uint4*)(ap + 40);
      bf1.a = *(const uint4*)(bp + 512);
      bf1.b = *(const uint4*)(bp + 520);
      z0 = wmma_bf16(frag_cast(af0), frag_cast(bf0), z0);
      z1 = wmma_bf16(frag_cast(af1), frag_cast(bf1), z1);
      ap += 64;
      bp += 1024;
    }
    const u16t* pp = &Ps[fo * 32 + bsub * 16 + half * 8]; // 8 rows of psi (b128)
    uint4 pv = *(const uint4*)pp;
    const u16t* pu = (const u16t*)&pv;
    for (int q = 0; q < 8; ++q) score[q] += bf2f(pu[q]) * (z0[q] + z1[q]);
  }

  const int scol = stile * 16 + row;
  float* dst = spart + (size_t)blockIdx.z * 131072;
  for (int q = 0; q < 8; ++q) {
    int brow = b0 + bsub * 16 + half * 8 + q;
    dst[(size_t)brow * 128 + scol] = score[q];
  }
}

// Sum the 4 f-chunk partials + psi.b2 bias, then DDPM update of xt in place.
__global__ void xt_update_kernel(const float* __restrict__ spart,
                                 const float* __restrict__ sbias,
                                 const float* __restrict__ sched,
                                 const float* __restrict__ noise,
                                 float* __restrict__ xt, int t, int istep) {
  int i = blockIdx.x * 256 + threadIdx.x;    // 0..131071
  float sc = sbias[i] + spart[i] + spart[131072 + i] +
             spart[262144 + i] + spart[393216 + i];
  float c1 = sched[t * 4 + 0], c2 = sched[t * 4 + 1], sg = sched[t * 4 + 2];
  float x = xt[i];
  float nz = noise[(size_t)istep * 131072 + i];
  float nx = (x - c1 * sc) * c2 + sg * nz;
  xt[i] = fminf(1.0f, fmaxf(-1.0f, nx));
}

// ---------------------------------------------------------------------------
extern "C" void kernel_launch(void* const* d_in, const int* in_sizes, int n_in,
                              void* d_out, int out_size, void* d_ws, size_t ws_size,
                              hipStream_t stream) {
  (void)in_sizes; (void)n_in; (void)out_size; (void)ws_size;
  const float* state   = (const float*)d_in[0];
  const float* action  = (const float*)d_in[1];
  const float* xt_init = (const float*)d_in[2];
  const float* noise   = (const float*)d_in[3];
  const float* P[24];
  for (int i = 0; i < 24; ++i) P[i] = (const float*)d_in[4 + i];
  const float *s_w1=P[0],*s_b1=P[1],*s_w2=P[2],*s_b2=P[3];
  const float *a_w1=P[4],*a_b1=P[5],*a_w2=P[6],*a_b2=P[7];
  const float *t_w1=P[8],*t_b1=P[9],*t_w2=P[10],*t_b2=P[11];
  const float *p_w0=P[12],*p_b0=P[13],*p_w1=P[14],*p_b1=P[15],*p_w2=P[16],*p_b2=P[17];
  const float *z_w0=P[18],*z_b0=P[19],*z_w1=P[20],*z_b1=P[21],*z_w2=P[22],*z_b2=P[23];

  char* ws = (char*)d_ws;
  size_t off = 0;
  auto alloc = [&](size_t bytes) -> void* {
    off = (off + 255) & ~(size_t)255;
    void* p = ws + off;
    off += bytes;
    return p;
  };
  auto pack = [&](const float* W, int K, int N) -> u16t* {
    u16t* Wp = (u16t*)alloc((size_t)K * N * 2);
    long long total = (long long)K * N;
    int blocks = (int)((total + 255) / 256);
    pack_w_kernel<<<dim3(blocks), dim3(256), 0, stream>>>(W, Wp, K, N);
    return Wp;
  };

  // packed bf16 weights (z_w2p = 67MB, L2-resident on MI455X's 192MB L2)
  u16t* s_w1p = pack(s_w1, 128, 256);
  u16t* s_w2p = pack(s_w2, 256, 128);
  u16t* a_w1p = pack(a_w1, 64, 256);
  u16t* a_w2p = pack(a_w2, 256, 128);
  u16t* p_w0p = pack(p_w0, 256, 512);
  u16t* p_w1p = pack(p_w1, 512, 512);
  u16t* p_w2p = pack(p_w2, 512, 256);
  u16t* z_w0p = pack(z_w0, 256, 1024);
  u16t* z_w1p = pack(z_w1, 1024, 1024);
  u16t* z_w2p = pack(z_w2, 1024, 32768);
  u16t* b2p   = pack(z_b2, 256, 128);     // z_b2 viewed as [256,128]

  u16t*  state_bf  = (u16t*)alloc((size_t)1024 * 128 * 2);
  u16t*  action_bf = (u16t*)alloc((size_t)1024 * 64 * 2);
  u16t*  t0  = (u16t*)alloc((size_t)1024 * 256 * 2);
  u16t*  t1  = (u16t*)alloc((size_t)1024 * 256 * 2);
  u16t*  x0  = (u16t*)alloc((size_t)1024 * 256 * 2);
  u16t*  t2  = (u16t*)alloc((size_t)1024 * 512 * 2);
  u16t*  t3  = (u16t*)alloc((size_t)1024 * 512 * 2);
  u16t*  psi_bf = (u16t*)alloc((size_t)1024 * 256 * 2);
  float* sbias  = (float*)alloc((size_t)1024 * 128 * 4);
  float* spart  = (float*)alloc((size_t)4 * 1024 * 128 * 4);
  float* sched  = (float*)alloc(256 * 4);
  float* zeros  = (float*)alloc(128 * 4);
  u16t*  tff    = (u16t*)alloc((size_t)50 * 128 * 2);
  float* xt     = (float*)alloc((size_t)1024 * 128 * 4);
  u16t*  xin    = (u16t*)alloc((size_t)1024 * 256 * 2);
  u16t*  h0     = (u16t*)alloc((size_t)1024 * 1024 * 2);
  u16t*  h1     = (u16t*)alloc((size_t)1024 * 1024 * 2);

  auto gemm = [&](const u16t* A, const u16t* Wp, const float* bias, void* out,
                  int K, int N, int ldo, int col_off, bool domish, bool outf32) {
    dim3 g(16, (unsigned)(N / 64));
    if (domish)
      gemm_kernel<true, false><<<g, 128, 0, stream>>>(A, Wp, bias, out, K, N, ldo, col_off);
    else if (outf32)
      gemm_kernel<false, true><<<g, 128, 0, stream>>>(A, Wp, bias, out, K, N, ldo, col_off);
    else
      gemm_kernel<false, false><<<g, 128, 0, stream>>>(A, Wp, bias, out, K, N, ldo, col_off);
  };

  sched_kernel<<<1, 128, 0, stream>>>(sched, zeros);
  tff_kernel<<<50, 256, 0, stream>>>(t_w1, t_b1, t_w2, t_b2, tff);
  cvt_bf16_kernel<<<(1024 * 128) / 256, 256, 0, stream>>>(state, state_bf, 1024 * 128);
  cvt_bf16_kernel<<<(1024 * 64) / 256, 256, 0, stream>>>(action, action_bf, 1024 * 64);
  (void)hipMemcpyAsync(xt, xt_init, (size_t)1024 * 128 * 4,
                       hipMemcpyDeviceToDevice, stream);

  // ψ branch (once): state/action MLPs -> concat -> 256->512->512->256
  gemm(state_bf,  s_w1p, s_b1, t0,     128, 256, 256, 0,   true,  false);
  gemm(t0,        s_w2p, s_b2, x0,     256, 128, 256, 0,   false, false);
  gemm(action_bf, a_w1p, a_b1, t1,      64, 256, 256, 0,   true,  false);
  gemm(t1,        a_w2p, a_b2, x0,     256, 128, 256, 128, false, false);
  gemm(x0,        p_w0p, p_b0, t2,     256, 512, 512, 0,   true,  false);
  gemm(t2,        p_w1p, p_b1, t3,     512, 512, 512, 0,   true,  false);
  gemm(t3,        p_w2p, p_b2, psi_bf, 512, 256, 256, 0,   false, false);
  // step-invariant bias term: sbias[b,s] = Σ_f psi[b,f] * z_b2[f,s]
  gemm(psi_bf, b2p, zeros, sbias, 256, 128, 128, 0, false, true);

  const size_t FUSED_LDS = (size_t)(32 * 1032 + 64 * 32) * sizeof(u16t); // ~69KB
  for (int i = 0; i < 50; ++i) {
    int t = 49 - i;                      // scan: t descends, noise[i] in order
    build_xin_kernel<<<1024, 256, 0, stream>>>(xt, tff, t, xin);
    gemm(xin, z_w0p, z_b0, h0,  256, 1024, 1024, 0, true, false);
    gemm(h0,  z_w1p, z_b1, h1, 1024, 1024, 1024, 0, true, false);
    fused_score_kernel<<<dim3(32, 2, 4), 256, FUSED_LDS, stream>>>(
        h1, z_w2p, psi_bf, spart);
    xt_update_kernel<<<512, 256, 0, stream>>>(spart, sbias, sched, noise, xt, t, i);
  }
  (void)hipMemcpyAsync(d_out, xt, (size_t)1024 * 128 * 4,
                       hipMemcpyDeviceToDevice, stream);
}
